// TransformerBlock_86870008529323
// MI455X (gfx1250) — matrix operand order repaired and checked by reading, not compiled, not run on hardware
//
#include <hip/hip_runtime.h>
#include <hip/hip_bf16.h>
#include <math.h>

#define DIM      768
#define SEQ      1024
#define BATCH    8
#define NTOK     (BATCH * SEQ)   // 8192
#define HEADS    12
#define HEAD_DIM 64
#define HIDDEN   3072

typedef __attribute__((ext_vector_type(16))) __bf16 v16bf;
typedef __attribute__((ext_vector_type(8)))  float  v8f;
typedef unsigned short bf16_t;   // raw bf16 bits in memory

// Feature detection (compile-safe fallback to load+ds_store path)
#if __has_builtin(__builtin_amdgcn_tensor_load_to_lds)
#define HAVE_TDM 1
#else
#define HAVE_TDM 0
#endif

// ---------- bf16 helpers (bit-exact RNE) ------------------------------------
static __device__ __forceinline__ bf16_t f2bf(float f) {
  unsigned int u = __builtin_bit_cast(unsigned int, f);
  unsigned int r = (u + 0x7FFFu + ((u >> 16) & 1u)) >> 16;
  return (bf16_t)r;
}

union FragAB {        // one WMMA 16-bit A/B operand: 16 bf16 = 8 VGPRs
  uint4 q[2];
  v16bf v;
};

static __device__ __forceinline__ v8f vzero8() {
  v8f z;
#pragma unroll
  for (int i = 0; i < 8; ++i) z[i] = 0.0f;
  return z;
}

static __device__ __forceinline__ v8f wmma_bf16(const FragAB& a, const FragAB& b, v8f c) {
  // D = A(16x32 bf16) * B(32x16 bf16) + C(16x16 f32)
  return __builtin_amdgcn_wmma_f32_16x16x32_bf16(false, a.v, false, b.v,
                                                 (short)0, c, false, false);
}

#if HAVE_TDM
// ---------------- Tensor Data Mover descriptor builders (ISA §8) ------------
typedef __attribute__((ext_vector_type(4))) unsigned int u32x4;
typedef __attribute__((ext_vector_type(4))) int          i32x4;
typedef __attribute__((ext_vector_type(8))) int          i32x8;
union D0u { unsigned long long q[2]; u32x4 v; };
union D1u { unsigned long long q[4]; i32x8 v; };

// group0: count=1 | lds_addr[63:32] | global_addr[120:64] | type=2 [127:126]
static __device__ __forceinline__ u32x4 tdm_g0(unsigned lds_addr,
                                               unsigned long long gaddr) {
  D0u d;
  d.q[0] = 1ull | ((unsigned long long)lds_addr << 32);
  d.q[1] = (gaddr & 0x1FFFFFFFFFFFFFFull) | (2ull << 62);
  return d.v;
}
// group1: 2-byte elems, tile 32(K) x 128(rows), pad 4 dwords after every
// 16 dwords (64B row) -> 80B LDS row stride (matches As/Bs layout).
static __device__ __forceinline__ i32x8 tdm_g1(unsigned tensor_d0, unsigned tensor_d1,
                                               unsigned long long stride_d0) {
  D1u d;
  unsigned w0 = (1u << 16)        // data_size = 2 bytes
              | (1u << 20)        // pad_enable
              | (3u << 22)        // pad_interval: 16 dwords
              | (3u << 25);       // pad_amount: 4 dwords
  d.q[0] = (unsigned long long)w0 |
           ((unsigned long long)(tensor_d0 & 0xFFFFu) << 48);   // dim0[15:0]
  d.q[1] = ((unsigned long long)(tensor_d0 >> 16) & 0xFFFFull)  // dim0[31:16]
         | ((unsigned long long)tensor_d1 << 16)                // dim1 [111:80]
         | (32ull << 48);                                       // tile_dim0=32
  d.q[2] = 128ull                                               // tile_dim1=128
         | ((stride_d0 & 0xFFFFFFFFull) << 32);                 // stride[31:0]
  d.q[3] = (stride_d0 >> 32) & 0xFFFFull;                       // stride[47:32]
  return d.v;
}
static __device__ __forceinline__ void tdm_load(u32x4 g0, i32x8 g1) {
  i32x4 z4 = {0, 0, 0, 0};
#if defined(__clang_major__) && (__clang_major__ >= 23)
  i32x8 z8 = {0, 0, 0, 0, 0, 0, 0, 0};
  __builtin_amdgcn_tensor_load_to_lds(g0, g1, z4, z4, z8, 0);
#else
  __builtin_amdgcn_tensor_load_to_lds(g0, g1, z4, z4, 0);
#endif
}
#endif  // HAVE_TDM

// ---------------------------------------------------------------------------
// fp32 -> bf16 weight conversion with transpose: out[n*K+k] = in[k*N+n]
// (one-time cost; makes both GEMM tiles identical row-major 2D copies)
// ---------------------------------------------------------------------------
__global__ __launch_bounds__(256) void convert_transpose_bf16_kernel(
    const float* __restrict__ in, bf16_t* __restrict__ outT, int K, int N) {
  int id = blockIdx.x * 256 + threadIdx.x;
  if (id >= K * N) return;
  int n = id / K, k = id % K;
  outT[id] = f2bf(in[(size_t)k * N + n]);
}

// ---------------------------------------------------------------------------
// LayerNorm over last dim (768), one row per block, bf16 output
// ---------------------------------------------------------------------------
__global__ __launch_bounds__(256) void layernorm_bf16_kernel(
    const float* __restrict__ x, const float* __restrict__ g,
    const float* __restrict__ b, bf16_t* __restrict__ out) {
  __shared__ float r1[256];
  __shared__ float r2[256];
  const int row = blockIdx.x;
  const int tid = threadIdx.x;
  const float* xr = x + (size_t)row * DIM;
  float v0 = xr[tid], v1 = xr[tid + 256], v2 = xr[tid + 512];
  r1[tid] = v0 + v1 + v2;
  r2[tid] = v0 * v0 + v1 * v1 + v2 * v2;
  __syncthreads();
  for (int off = 128; off > 0; off >>= 1) {
    if (tid < off) { r1[tid] += r1[tid + off]; r2[tid] += r2[tid + off]; }
    __syncthreads();
  }
  const float mean = r1[0] * (1.0f / DIM);
  const float var  = r2[0] * (1.0f / DIM) - mean * mean;
  const float rstd = rsqrtf(var + 1e-5f);
  bf16_t* orow = out + (size_t)row * DIM;
#pragma unroll
  for (int i = tid; i < DIM; i += 256)
    orow[i] = f2bf((xr[i] - mean) * rstd * g[i] + b[i]);
}

// ---------------------------------------------------------------------------
// bf16 WMMA GEMM: C[M,N] = A[M,K] * Bt[N,K]^T (+bias) (+epilogue)
//   mode 0: store bf16     mode 1: GELU(exact) -> bf16
//   mode 2: + residual f32 -> store f32
// 256 thr (8 waves, 2x4), 128x128 tile, BK=32.
// A tile: TDM DMA issued by wave 0; B tile: TDM DMA issued by wave 1
// (tensor_load_to_lds, TENSORcnt). Fallback: global_load_b128+ds_store_b128.
// LDS rows padded to 80B (TDM pad_interval/pad_amount reproduce this).
// ---------------------------------------------------------------------------
__global__ __launch_bounds__(256) void gemm_bf16_kernel(
    const bf16_t* __restrict__ A, const bf16_t* __restrict__ Bt,
    const float* __restrict__ bias, const float* __restrict__ residual,
    void* __restrict__ out, int M, int N, int K, int mode) {
  __shared__ __align__(16) bf16_t As[128 * 40];
  __shared__ __align__(16) bf16_t Bs[128 * 40];

  const int tid   = threadIdx.x;
  const int lane  = tid & 31;
  const int wave  = tid >> 5;
  const int laneN = lane & 15;
  const int lh    = (lane >> 4) & 1;   // lane-half (ISA fragment split)
  const int wm    = wave >> 2;         // 0..1 -> 64 rows
  const int wn    = wave & 3;          // 0..3 -> 32 cols
  const int block_m = blockIdx.y * 128;
  const int block_n = blockIdx.x * 128;

  v8f acc[4][2];
#pragma unroll
  for (int mt = 0; mt < 4; ++mt)
#pragma unroll
    for (int nt = 0; nt < 2; ++nt) acc[mt][nt] = vzero8();

#if HAVE_TDM
  const i32x8 g1A = tdm_g1((unsigned)K, (unsigned)M, (unsigned long long)K);
  const i32x8 g1B = tdm_g1((unsigned)K, (unsigned)N, (unsigned long long)K);
  const unsigned ldsA = (unsigned)(size_t)(void*)&As[0];
  const unsigned ldsB = (unsigned)(size_t)(void*)&Bs[0];
#endif

  for (int k0 = 0; k0 < K; k0 += 32) {
#if HAVE_TDM
    // ---- stage A tile 128x32 via Tensor Data Mover (wave 0) ----
    if (wave == 0) {
      unsigned long long ga =
          (unsigned long long)(size_t)A + 2ull * ((size_t)block_m * K + k0);
      tdm_load(tdm_g0(ldsA, ga), g1A);   // tensor_load_to_lds
    }
    // ---- stage B tile 128x32 via Tensor Data Mover (wave 1) ----
    if (wave == 1) {
      unsigned long long gb =
          (unsigned long long)(size_t)Bt + 2ull * ((size_t)block_n * K + k0);
      tdm_load(tdm_g0(ldsB, gb), g1B);   // tensor_load_to_lds
    }
#else
#pragma unroll
    for (int it = 0; it < 2; ++it) {
      int idx = it * 2048 + tid * 8;
      int r = idx >> 5, c = idx & 31;
      *(uint4*)&As[r * 40 + c] =
          *(const uint4*)&A[(size_t)(block_m + r) * K + (k0 + c)];
      *(uint4*)&Bs[r * 40 + c] =
          *(const uint4*)&Bt[(size_t)(block_n + r) * K + (k0 + c)];
    }
#endif
    // next K-tile prefetch (global_prefetch_b8), one block per step
    if (k0 + 32 < K) {
      __builtin_prefetch(&A[(size_t)(block_m + (tid >> 1)) * K + k0 + 32], 0, 0);
      __builtin_prefetch(&Bt[(size_t)(block_n + (tid >> 1)) * K + k0 + 32], 0, 0);
    }
#if HAVE_TDM
    if (wave < 2) __builtin_amdgcn_s_wait_tensorcnt(0);
#endif
    __syncthreads();

    // ---- fragments: 2x ds_load_b128 each (contiguous k = lh*16 + 0..15 in both A and B) ----
    FragAB af[4], bf_[2];
#pragma unroll
    for (int mt = 0; mt < 4; ++mt) {
      const bf16_t* s = &As[(wm * 64 + mt * 16 + laneN) * 40 + lh * 16];
      af[mt].q[0] = *(const uint4*)(s);        // k = lh*16 + 0..7
      af[mt].q[1] = *(const uint4*)(s + 8);    // k = lh*16 + 8..15
    }
#pragma unroll
    for (int nt = 0; nt < 2; ++nt) {
      const bf16_t* s = &Bs[(wn * 32 + nt * 16 + laneN) * 40 + lh * 16];
      bf_[nt].q[0] = *(const uint4*)(s);       // k = lh*16 + 0..7
      bf_[nt].q[1] = *(const uint4*)(s + 8);   // k = lh*16 + 8..15
    }
#pragma unroll
    for (int mt = 0; mt < 4; ++mt)
#pragma unroll
      for (int nt = 0; nt < 2; ++nt)
        acc[mt][nt] = wmma_bf16(af[mt], bf_[nt], acc[mt][nt]);
    __syncthreads();
  }

  // ---- epilogue: C layout VGPR r -> row r (lanes 0-15) / r+8 (lanes 16-31)
#pragma unroll
  for (int mt = 0; mt < 4; ++mt) {
#pragma unroll
    for (int nt = 0; nt < 2; ++nt) {
#pragma unroll
      for (int r = 0; r < 8; ++r) {
        int grow = block_m + wm * 64 + mt * 16 + r + lh * 8;
        int gcol = block_n + wn * 32 + nt * 16 + laneN;
        float v = acc[mt][nt][r];
        if (bias) v += bias[gcol];
        if (mode == 2) {
          v += residual[(size_t)grow * N + gcol];
          ((float*)out)[(size_t)grow * N + gcol] = v;
        } else {
          if (mode == 1) v = 0.5f * v * (1.0f + erff(v * 0.70710678118f));
          ((bf16_t*)out)[(size_t)grow * N + gcol] = f2bf(v);
        }
      }
    }
  }
}

// ---------------------------------------------------------------------------
// V transpose: qkv[b,n,2,h,d] -> Vt[b,h,d,n] (so P*V B-fragments are contiguous)
// ---------------------------------------------------------------------------
__global__ __launch_bounds__(256) void transpose_v_kernel(
    const bf16_t* __restrict__ qkv, bf16_t* __restrict__ vt) {
  int id = blockIdx.x * 256 + threadIdx.x;   // id = ((b*H + h)*D + d)*SEQ + n
  int n = id & (SEQ - 1);
  int t = id >> 10;
  int d = t & (HEAD_DIM - 1);
  t >>= 6;
  int h = t % HEADS;
  int b = t / HEADS;
  vt[id] = qkv[(size_t)(b * SEQ + n) * (3 * DIM) + 2 * DIM + h * HEAD_DIM + d];
}

// ---------------------------------------------------------------------------
// Flash attention: WG = 4 waves, each wave owns 16 query rows of one (b,h).
// Per 32-key step: S = Q*K^T (4 WMMAs), online softmax, P via wave-private
// LDS relayout (s_wait_dscnt), O += P*V (4 WMMAs).
// ---------------------------------------------------------------------------
__global__ __launch_bounds__(128) void attention_kernel(
    const bf16_t* __restrict__ qkv, const bf16_t* __restrict__ vt,
    bf16_t* __restrict__ out) {
  __shared__ __align__(16) bf16_t Plds[4][16 * 40];

  const int tid   = threadIdx.x;
  const int lane  = tid & 31;
  const int wave  = tid >> 5;
  const int laneN = lane & 15;
  const int lh    = (lane >> 4) & 1;
  const int qblk  = blockIdx.x;            // 0..15
  const int h     = blockIdx.y;            // 0..11
  const int b     = blockIdx.z;            // 0..7
  const int q0    = qblk * 64 + wave * 16; // first query row of this wave

  // Q fragments (A operand), resident for the whole key loop
  FragAB qf[2];
  {
    const bf16_t* qp =
        qkv + (size_t)(b * SEQ + q0 + laneN) * (3 * DIM) + h * HEAD_DIM;
#pragma unroll
    for (int ks = 0; ks < 2; ++ks) {
      const bf16_t* p = qp + ks * 32 + lh * 16;
      qf[ks].q[0] = *(const uint4*)(p);
      qf[ks].q[1] = *(const uint4*)(p + 8);
    }
  }

  v8f oacc[4];
#pragma unroll
  for (int dt = 0; dt < 4; ++dt) oacc[dt] = vzero8();
  float mrow[8], lrow[8];
#pragma unroll
  for (int r = 0; r < 8; ++r) { mrow[r] = -1e30f; lrow[r] = 0.0f; }

  bf16_t* pl = &Plds[wave][0];
  const float scale = 0.125f;  // HEAD_DIM^-0.5

  for (int j = 0; j < SEQ; j += 32) {
    // ---- S = Q * K^T for 32 keys (two 16-col tiles) ----
    v8f sacc[2];
    sacc[0] = vzero8(); sacc[1] = vzero8();
#pragma unroll
    for (int nt = 0; nt < 2; ++nt) {
      const bf16_t* kp = qkv +
          (size_t)(b * SEQ + j + nt * 16 + laneN) * (3 * DIM) + DIM + h * HEAD_DIM;
#pragma unroll
      for (int ks = 0; ks < 2; ++ks) {
        FragAB kf;
        const bf16_t* p = kp + ks * 32 + lh * 16;
        kf.q[0] = *(const uint4*)(p);
        kf.q[1] = *(const uint4*)(p + 8);
        sacc[nt] = wmma_bf16(qf[ks], kf, sacc[nt]);
      }
    }

    // ---- online softmax (row stats via 16-lane xor-shuffles) ----
#pragma unroll
    for (int r = 0; r < 8; ++r) {
      float s0 = sacc[0][r] * scale;
      float s1 = sacc[1][r] * scale;
      float mx = fmaxf(s0, s1);
#pragma unroll
      for (int off = 1; off < 16; off <<= 1)
        mx = fmaxf(mx, __shfl_xor(mx, off, 16));
      float mnew  = fmaxf(mrow[r], mx);
      float alpha = __expf(mrow[r] - mnew);
      float p0 = __expf(s0 - mnew);
      float p1 = __expf(s1 - mnew);
      float ps = p0 + p1;
#pragma unroll
      for (int off = 1; off < 16; off <<= 1)
        ps += __shfl_xor(ps, off, 16);
      lrow[r] = lrow[r] * alpha + ps;
      mrow[r] = mnew;
#pragma unroll
      for (int dt = 0; dt < 4; ++dt) oacc[dt][r] *= alpha;
      int prow = r + lh * 8;               // C-layout row this register holds
      pl[prow * 40 + laneN]      = f2bf(p0);
      pl[prow * 40 + 16 + laneN] = f2bf(p1);
    }
    // wave-private LDS region: DS ops in-order per wave, drain DScnt
    asm volatile("s_wait_dscnt 0x0" ::: "memory");

    // ---- relayout P(16x32) as A fragment, O += P * V ----
    FragAB pf;
    {
      const bf16_t* p = &pl[laneN * 40 + lh * 16];
      pf.q[0] = *(const uint4*)(p);
      pf.q[1] = *(const uint4*)(p + 8);
    }
#pragma unroll
    for (int dt = 0; dt < 4; ++dt) {
      FragAB vf;
      const bf16_t* p = vt +
          ((size_t)((b * HEADS + h) * HEAD_DIM) + dt * 16 + laneN) * SEQ +
          j + lh * 16;
      vf.q[0] = *(const uint4*)(p);
      vf.q[1] = *(const uint4*)(p + 8);
      oacc[dt] = wmma_bf16(pf, vf, oacc[dt]);
    }
  }

  // ---- normalize and emit bf16 [token, h*64+d] ----
#pragma unroll
  for (int dt = 0; dt < 4; ++dt) {
#pragma unroll
    for (int r = 0; r < 8; ++r) {
      int token = b * SEQ + q0 + r + lh * 8;
      int col   = h * HEAD_DIM + dt * 16 + laneN;
      out[(size_t)token * DIM + col] = f2bf(oacc[dt][r] / lrow[r]);
    }
  }
}

// ---------------------------------------------------------------------------
extern "C" void kernel_launch(void* const* d_in, const int* in_sizes, int n_in,
                              void* d_out, int out_size, void* d_ws, size_t ws_size,
                              hipStream_t stream) {
  (void)in_sizes; (void)n_in; (void)out_size; (void)ws_size;
  const float* x      = (const float*)d_in[0];
  const float* n1g    = (const float*)d_in[1];
  const float* n1b    = (const float*)d_in[2];
  const float* qkv_w  = (const float*)d_in[3];
  const float* qkv_b  = (const float*)d_in[4];
  const float* proj_w = (const float*)d_in[5];
  const float* proj_b = (const float*)d_in[6];
  const float* n2g    = (const float*)d_in[7];
  const float* n2b    = (const float*)d_in[8];
  const float* fc1_w  = (const float*)d_in[9];
  const float* fc1_b  = (const float*)d_in[10];
  const float* fc2_w  = (const float*)d_in[11];
  const float* fc2_b  = (const float*)d_in[12];

  char* ws = (char*)d_ws;
  size_t off = 0;
  auto alloc = [&](size_t bytes) -> void* {
    void* p = ws + off;
    off += (bytes + 255) & ~(size_t)255;
    return p;
  };
  bf16_t* xn    = (bf16_t*)alloc((size_t)NTOK * DIM * 2);
  bf16_t* qkvw  = (bf16_t*)alloc((size_t)DIM * 3 * DIM * 2);   // [2304,768]
  bf16_t* projw = (bf16_t*)alloc((size_t)DIM * DIM * 2);       // [768,768]
  bf16_t* fc1w  = (bf16_t*)alloc((size_t)DIM * HIDDEN * 2);    // [3072,768]
  bf16_t* fc2w  = (bf16_t*)alloc((size_t)HIDDEN * DIM * 2);    // [768,3072]
  bf16_t* qkvo  = (bf16_t*)alloc((size_t)NTOK * 3 * DIM * 2);
  bf16_t* vtb   = (bf16_t*)alloc((size_t)NTOK * DIM * 2);
  bf16_t* attno = (bf16_t*)alloc((size_t)NTOK * DIM * 2);
  float*  x1    = (float*) alloc((size_t)NTOK * DIM * 4);
  bf16_t* hnorm = (bf16_t*)alloc((size_t)NTOK * DIM * 2);
  bf16_t* h1    = (bf16_t*)alloc((size_t)NTOK * HIDDEN * 2);

  // weights fp32 -> bf16, transposed to [N,K]
  convert_transpose_bf16_kernel<<<(DIM * 3 * DIM + 255) / 256, 256, 0, stream>>>(
      qkv_w, qkvw, DIM, 3 * DIM);
  convert_transpose_bf16_kernel<<<(DIM * DIM + 255) / 256, 256, 0, stream>>>(
      proj_w, projw, DIM, DIM);
  convert_transpose_bf16_kernel<<<(DIM * HIDDEN + 255) / 256, 256, 0, stream>>>(
      fc1_w, fc1w, DIM, HIDDEN);
  convert_transpose_bf16_kernel<<<(HIDDEN * DIM + 255) / 256, 256, 0, stream>>>(
      fc2_w, fc2w, HIDDEN, DIM);

  // LN1 -> bf16
  layernorm_bf16_kernel<<<NTOK, 256, 0, stream>>>(x, n1g, n1b, xn);

  // qkv = xn @ qkv_w + b  -> bf16 [8192, 2304]
  gemm_bf16_kernel<<<dim3(3 * DIM / 128, NTOK / 128), 256, 0, stream>>>(
      xn, qkvw, qkv_b, nullptr, qkvo, NTOK, 3 * DIM, DIM, 0);

  // V -> Vt[b,h,d,n]
  transpose_v_kernel<<<(size_t)NTOK * DIM / 256, 256, 0, stream>>>(qkvo, vtb);

  // flash attention -> bf16 [8192, 768]
  attention_kernel<<<dim3(SEQ / 64, HEADS, BATCH), 128, 0, stream>>>(qkvo, vtb, attno);

  // x1 = x + attn @ proj_w + b   (f32)
  gemm_bf16_kernel<<<dim3(DIM / 128, NTOK / 128), 256, 0, stream>>>(
      attno, projw, proj_b, x, x1, NTOK, DIM, DIM, 2);

  // LN2 -> bf16
  layernorm_bf16_kernel<<<NTOK, 256, 0, stream>>>(x1, n2g, n2b, hnorm);

  // h1 = gelu(hnorm @ fc1_w + b) -> bf16 [8192, 3072]
  gemm_bf16_kernel<<<dim3(HIDDEN / 128, NTOK / 128), 256, 0, stream>>>(
      hnorm, fc1w, fc1_b, nullptr, h1, NTOK, HIDDEN, DIM, 1);

  // out = x1 + h1 @ fc2_w + b   (f32)
  gemm_bf16_kernel<<<dim3(DIM / 128, NTOK / 128), 256, 0, stream>>>(
      h1, fc2w, fc2_b, x1, (float*)d_out, NTOK, DIM, HIDDEN, 2);
}
